// ConcatModel_17282948399385
// MI455X (gfx1250) — matrix operand order repaired and checked by reading, not compiled, not run on hardware
//
#include <hip/hip_runtime.h>

// ---------------------------------------------------------------------------
// Problem dims (from reference): B=8192, E=100, D=300, P=100, H=20
// ---------------------------------------------------------------------------
#define BDIM 8192
#define ED   100
#define DD   300
#define PD   100
#define HD   20
#define FD   (2 * PD + 2)   // 202 features
#define DPAD 320            // D padded to 10 K-steps of 32 (bf16 WMMA K)
#define PPAD 128            // P padded to 8 N-tiles of 16
#define MT   32             // batch rows per block (2 WMMA M-tiles)

typedef __attribute__((ext_vector_type(4)))  unsigned int u32x4;
typedef __attribute__((ext_vector_type(16))) __bf16       v16bf;
typedef __attribute__((ext_vector_type(8)))  float        v8f;

union FragBF {
    u32x4 q[2];
    v16bf v;
};

// float -> bf16 (round to nearest even), stored as raw u16
static __device__ __forceinline__ unsigned short f2bf(float f) {
    unsigned int u = __float_as_uint(f);
    u += 0x7FFFu + ((u >> 16) & 1u);
    return (unsigned short)(u >> 16);
}

// ---------------------------------------------------------------------------
// Prep 1: xs = bf16(x+y), xd = bf16(x-y), padded [B, DPAD]
// ---------------------------------------------------------------------------
__global__ __launch_bounds__(256)
void prep_xy_kernel(const float* __restrict__ x, const float* __restrict__ y,
                    unsigned short* __restrict__ xs, unsigned short* __restrict__ xd) {
    size_t i = (size_t)blockIdx.x * 256 + threadIdx.x;
    if (i >= (size_t)BDIM * DPAD) return;
    int    k = (int)(i % DPAD);
    size_t b = i / DPAD;
    unsigned short s = 0, d = 0;
    if (k < DD) {
        float xv = x[b * DD + k];
        float yv = y[b * DD + k];
        s = f2bf(xv + yv);
        d = f2bf(xv - yv);
    }
    xs[i] = s;
    xd[i] = d;
}

// ---------------------------------------------------------------------------
// Prep 2: w1t[e][p][k] = bf16(W1[e][k][p]), padded [E, PPAD, DPAD]
// (p-major so each lane's B-fragment is two runs of 8 contiguous K values)
// ---------------------------------------------------------------------------
__global__ __launch_bounds__(256)
void prep_w1_kernel(const float* __restrict__ w1, unsigned short* __restrict__ w1t) {
    size_t i = (size_t)blockIdx.x * 256 + threadIdx.x;
    if (i >= (size_t)ED * PPAD * DPAD) return;
    int    k = (int)(i % DPAD);
    size_t t = i / DPAD;
    int    p = (int)(t % PPAD);
    int    e = (int)(t / PPAD);
    unsigned short v = 0;
    if (k < DD && p < PD) v = f2bf(w1[((size_t)e * DD + k) * PD + p]);
    w1t[i] = v;
}

// ---------------------------------------------------------------------------
// Gating: w[b,:] = softmax((x[b]+y[b]) @ weights_all)   one block per row
// ---------------------------------------------------------------------------
__global__ __launch_bounds__(128)
void gate_kernel(const float* __restrict__ x, const float* __restrict__ y,
                 const float* __restrict__ wall, float* __restrict__ gate) {
    __shared__ float logits[ED];
    const int b   = blockIdx.x;
    const int tid = threadIdx.x;
    if (tid < ED) {
        float acc = 0.0f;
        const float* xr = x + (size_t)b * DD;
        const float* yr = y + (size_t)b * DD;
        for (int d = 0; d < DD; ++d) acc += (xr[d] + yr[d]) * wall[(size_t)d * ED + tid];
        logits[tid] = acc;
    }
    __syncthreads();
    if (tid < 32) {  // wave 0 does the softmax (wave-synchronous)
        float mx = -3.0e38f;
        for (int e = tid; e < ED; e += 32) mx = fmaxf(mx, logits[e]);
        for (int o = 16; o > 0; o >>= 1) mx = fmaxf(mx, __shfl_down(mx, o, 32));
        mx = __shfl(mx, 0, 32);
        float sm = 0.0f;
        for (int e = tid; e < ED; e += 32) sm += __expf(logits[e] - mx);
        for (int o = 16; o > 0; o >>= 1) sm += __shfl_down(sm, o, 32);
        sm = __shfl(sm, 0, 32);
        float inv = 1.0f / sm;
        for (int e = tid; e < ED; e += 32)
            gate[(size_t)b * ED + e] = __expf(logits[e] - mx) * inv;
    }
}

// ---------------------------------------------------------------------------
// Main: per (32-row batch tile, expert): u=(x+y)@W1+2b1, v=(x-y)@W1 via WMMA,
// then cosine + hidden MLP + out layer; writes w[b,e]*d[b,e].
// 4 waves; wave w owns P-tiles {2w, 2w+1} for both 16-row M-tiles.
// 8 v_wmma per K-step, 80 per wave; 12 b128 loads per K-step (1.5/wmma).
// ---------------------------------------------------------------------------
__global__ __launch_bounds__(128)
void expert_kernel(const unsigned short* __restrict__ xs,   // [B, DPAD] bf16
                   const unsigned short* __restrict__ xd,   // [B, DPAD] bf16
                   const unsigned short* __restrict__ w1t,  // [E, PPAD, DPAD] bf16
                   const float* __restrict__ bias1,         // [E, PD]
                   const float* __restrict__ hw,            // [E, FD, HD]
                   const float* __restrict__ hb,            // [E, HD]
                   const float* __restrict__ ow,            // [E, HD]
                   const float* __restrict__ ob,            // [E]
                   const float* __restrict__ tag,           // [B]
                   const float* __restrict__ gate,          // [B, E]
                   float* __restrict__ dcontrib)            // [B, E]
{
    const int b0   = blockIdx.x * MT;
    const int e    = blockIdx.y;
    const int tid  = threadIdx.x;
    const int wave = tid >> 5;
    const int lane = tid & 31;
    const int m    = lane & 15;   // column index within A-row / B-column tile
    const int kh   = lane >> 4;   // lane-half selector

    __shared__ float u_lds[MT][PPAD];
    __shared__ float v_lds[MT][PPAD];
    __shared__ float cos_lds[MT];
    __shared__ float red_lds[MT][4];

    // A-fragment sources: two 16-row M-tiles of xs/xd
    const unsigned short* aS0 = xs + (size_t)(b0 + m) * DPAD;
    const unsigned short* aS1 = xs + (size_t)(b0 + 16 + m) * DPAD;
    const unsigned short* aD0 = xd + (size_t)(b0 + m) * DPAD;
    const unsigned short* aD1 = xd + (size_t)(b0 + 16 + m) * DPAD;
    // B-fragment sources: wave's two P-tiles, lane's column
    const int p0 = wave * 32;
    const unsigned short* bB0 = w1t + ((size_t)e * PPAD + (p0 + m)) * DPAD;
    const unsigned short* bB1 = w1t + ((size_t)e * PPAD + (p0 + 16 + m)) * DPAD;

    v8f aU00 = {}, aU01 = {}, aU10 = {}, aU11 = {};
    v8f aV00 = {}, aV01 = {}, aV10 = {}, aV11 = {};

    for (int k0 = 0; k0 < DPAD; k0 += 32) {
        FragBF s0, s1, d0, d1, f0, f1;
        // A layout: element i -> K = (i>>3)*16 + kh*8 + (i&7)
        s0.q[0] = *(const u32x4*)(aS0 + k0 + kh * 8);
        s0.q[1] = *(const u32x4*)(aS0 + k0 + 16 + kh * 8);
        s1.q[0] = *(const u32x4*)(aS1 + k0 + kh * 8);
        s1.q[1] = *(const u32x4*)(aS1 + k0 + 16 + kh * 8);
        d0.q[0] = *(const u32x4*)(aD0 + k0 + kh * 8);
        d0.q[1] = *(const u32x4*)(aD0 + k0 + 16 + kh * 8);
        d1.q[0] = *(const u32x4*)(aD1 + k0 + kh * 8);
        d1.q[1] = *(const u32x4*)(aD1 + k0 + 16 + kh * 8);
        // B layout: element i -> K = (i>>3)*16 + kh*8 + (i&7)
        f0.q[0] = *(const u32x4*)(bB0 + k0 + kh * 8);
        f0.q[1] = *(const u32x4*)(bB0 + k0 + 16 + kh * 8);
        f1.q[0] = *(const u32x4*)(bB1 + k0 + kh * 8);
        f1.q[1] = *(const u32x4*)(bB1 + k0 + 16 + kh * 8);
        // prefetch next K-chunk of the weight stream
        __builtin_prefetch(bB0 + k0 + 32, 0, 1);
        __builtin_prefetch(bB1 + k0 + 32, 0, 1);

        aU00 = __builtin_amdgcn_wmma_f32_16x16x32_bf16(false, s0.v, false, f0.v,
                                                       (short)0, aU00, false, false);
        aU01 = __builtin_amdgcn_wmma_f32_16x16x32_bf16(false, s0.v, false, f1.v,
                                                       (short)0, aU01, false, false);
        aU10 = __builtin_amdgcn_wmma_f32_16x16x32_bf16(false, s1.v, false, f0.v,
                                                       (short)0, aU10, false, false);
        aU11 = __builtin_amdgcn_wmma_f32_16x16x32_bf16(false, s1.v, false, f1.v,
                                                       (short)0, aU11, false, false);
        aV00 = __builtin_amdgcn_wmma_f32_16x16x32_bf16(false, d0.v, false, f0.v,
                                                       (short)0, aV00, false, false);
        aV01 = __builtin_amdgcn_wmma_f32_16x16x32_bf16(false, d0.v, false, f1.v,
                                                       (short)0, aV01, false, false);
        aV10 = __builtin_amdgcn_wmma_f32_16x16x32_bf16(false, d1.v, false, f0.v,
                                                       (short)0, aV10, false, false);
        aV11 = __builtin_amdgcn_wmma_f32_16x16x32_bf16(false, d1.v, false, f1.v,
                                                       (short)0, aV11, false, false);
    }

    // D layout: element r of acc -> row (r + 8*kh) [+16 for second M-tile],
    //           col = tile_base + m
    const int colA = p0 + m;
    const int colB = p0 + 16 + m;
    const bool vA = (colA < PD);
    const bool vB = (colB < PD);
    const float biasA = vA ? 2.0f * bias1[(size_t)e * PD + colA] : 0.0f;
    const float biasB = vB ? 2.0f * bias1[(size_t)e * PD + colB] : 0.0f;
#pragma unroll
    for (int r = 0; r < 8; ++r) {
        const int row0 = r + kh * 8;
        const int row1 = row0 + 16;
        u_lds[row0][colA] = vA ? (aU00[r] + biasA) : 0.0f;
        u_lds[row0][colB] = vB ? (aU01[r] + biasB) : 0.0f;
        u_lds[row1][colA] = vA ? (aU10[r] + biasA) : 0.0f;
        u_lds[row1][colB] = vB ? (aU11[r] + biasB) : 0.0f;
        v_lds[row0][colA] = vA ? aV00[r] : 0.0f;
        v_lds[row0][colB] = vB ? aV01[r] : 0.0f;
        v_lds[row1][colA] = vA ? aV10[r] : 0.0f;
        v_lds[row1][colB] = vB ? aV11[r] : 0.0f;
    }
    __syncthreads();

    // cosine from u = xp+yp, v = xp-yp:
    //   xp.yp = (|u|^2 - |v|^2)/4 ; |xp|^2 = (|u|^2 + 2u.v + |v|^2)/4 ; etc.
    if (tid < MT) {
        float su = 0.0f, sv = 0.0f, sd = 0.0f;
        for (int p = 0; p < PD; ++p) {
            float uu = u_lds[tid][p], vv = v_lds[tid][p];
            su += uu * uu; sv += vv * vv; sd += uu * vv;
        }
        const float xy = 0.25f * (su - sv);
        const float xx = 0.25f * (su + 2.0f * sd + sv);
        const float yy = 0.25f * (su - 2.0f * sd + sv);
        const float eps = 1e-8f;
        const float nx = fmaxf(sqrtf(fmaxf(xx, 0.0f)), eps);
        const float ny = fmaxf(sqrtf(fmaxf(yy, 0.0f)), eps);
        cos_lds[tid] = xy / (nx * ny);
    }
    __syncthreads();

    // hidden layer (202 -> 20, ReLU) + out layer; 4 threads per row
    const int row = tid >> 2;
    const int l4  = tid & 3;
    const int b   = b0 + row;
    const float csv = cos_lds[row];
    const float tg  = tag[b];
    const float* Wh = hw + (size_t)e * FD * HD;
    float dpart = 0.0f;
    for (int j = l4; j < HD; j += 4) {
        float acc = hb[(size_t)e * HD + j];
        acc += csv * Wh[j];
        for (int p = 0; p < PD; ++p) acc += u_lds[row][p] * Wh[(1 + p) * HD + j];
        for (int p = 0; p < PD; ++p) acc += fabsf(v_lds[row][p]) * Wh[(1 + PD + p) * HD + j];
        acc += tg * Wh[(1 + 2 * PD) * HD + j];
        acc = fmaxf(acc, 0.0f);
        dpart += acc * ow[(size_t)e * HD + j];
    }
    red_lds[row][l4] = dpart;
    __syncthreads();
    if (l4 == 0) {
        float s = red_lds[row][0] + red_lds[row][1] + red_lds[row][2] + red_lds[row][3];
        s += ob[e];
        dcontrib[(size_t)b * ED + e] = s * gate[(size_t)b * ED + e];
    }
}

// ---------------------------------------------------------------------------
// Final deterministic reduction over experts: out[b] = sum_e dcontrib[b,e]
// ---------------------------------------------------------------------------
__global__ __launch_bounds__(256)
void reduce_kernel(const float* __restrict__ dcontrib, float* __restrict__ out) {
    const int b    = blockIdx.x * 8 + (threadIdx.x >> 5);
    const int lane = threadIdx.x & 31;
    if (b >= BDIM) return;
    float s = 0.0f;
    for (int e = lane; e < ED; e += 32) s += dcontrib[(size_t)b * ED + e];
    for (int o = 16; o > 0; o >>= 1) s += __shfl_down(s, o, 32);
    if (lane == 0) out[b] = s;
}

// ---------------------------------------------------------------------------
// Launch
// ---------------------------------------------------------------------------
extern "C" void kernel_launch(void* const* d_in, const int* in_sizes, int n_in,
                              void* d_out, int out_size, void* d_ws, size_t ws_size,
                              hipStream_t stream) {
    const float* x    = (const float*)d_in[0];
    const float* y    = (const float*)d_in[1];
    const float* tag  = (const float*)d_in[2];
    const float* w1   = (const float*)d_in[3];
    const float* b1   = (const float*)d_in[4];
    const float* hw   = (const float*)d_in[5];
    const float* hb   = (const float*)d_in[6];
    const float* ow   = (const float*)d_in[7];
    const float* ob   = (const float*)d_in[8];
    const float* wall = (const float*)d_in[9];
    float* out = (float*)d_out;

    // workspace layout (bytes)
    const size_t XS_BYTES  = (size_t)BDIM * DPAD * 2;            // 5.24 MB
    const size_t W1T_BYTES = (size_t)ED * PPAD * DPAD * 2;       // 8.19 MB
    const size_t GE_BYTES  = (size_t)BDIM * ED * 4;              // 3.28 MB
    char* ws = (char*)d_ws;
    unsigned short* xs  = (unsigned short*)(ws);
    unsigned short* xd  = (unsigned short*)(ws + XS_BYTES);
    unsigned short* w1t = (unsigned short*)(ws + 2 * XS_BYTES);
    float* gate     = (float*)(ws + 2 * XS_BYTES + W1T_BYTES);
    float* dcontrib = (float*)(ws + 2 * XS_BYTES + W1T_BYTES + GE_BYTES);

    {
        const size_t n = (size_t)BDIM * DPAD;
        prep_xy_kernel<<<(unsigned)((n + 255) / 256), 256, 0, stream>>>(x, y, xs, xd);
    }
    {
        const size_t n = (size_t)ED * PPAD * DPAD;
        prep_w1_kernel<<<(unsigned)((n + 255) / 256), 256, 0, stream>>>(w1, w1t);
    }
    gate_kernel<<<BDIM, 128, 0, stream>>>(x, y, wall, gate);

    dim3 grid(BDIM / MT, ED);
    expert_kernel<<<grid, 128, 0, stream>>>(xs, xd, w1t, b1, hw, hb, ow, ob,
                                            tag, gate, dcontrib);

    reduce_kernel<<<BDIM / 8, 256, 0, stream>>>(dcontrib, out);
}
